// TH_LSTM_65592740545323
// MI455X (gfx1250) — matrix operand order repaired and checked by reading, not compiled, not run on hardware
//
#include <hip/hip_runtime.h>
#include <hip/hip_bf16.h>

// ---------------------------------------------------------------------------
// TH-LSTM for MI455X (gfx1250): bf16 WMMA GEMMs + f32 elementwise recurrence.
// B=512, T=128, D=256, H=512, MEMORY=64.
// ---------------------------------------------------------------------------

#define BB   512
#define TT   128
#define DD   256
#define HH   512
#define MM   64
#define G5H  2560   // 5*H

typedef __attribute__((ext_vector_type(16))) __bf16 v16bf;
typedef __attribute__((ext_vector_type(8)))  float  v8f;

#define WMMA_BF16(a, b, c) \
  __builtin_amdgcn_wmma_f32_16x16x32_bf16(false, (a), false, (b), (short)0, (c), false, false)

// -------------------------- small helpers ----------------------------------

__device__ __forceinline__ unsigned short f32_to_bf16(float f) {
  union { float f; unsigned u; } v; v.f = f;
  unsigned u = v.u;
  unsigned rb = 0x7FFFu + ((u >> 16) & 1u);   // round-to-nearest-even
  return (unsigned short)((u + rb) >> 16);
}

__device__ __forceinline__ float sigmoidf(float x) { return 1.f / (1.f + __expf(-x)); }

union BF8 { uint4 u; __bf16 h[8]; };

// A fragment: 16x32 bf16 (M x K). Row-major A (bf16), stride lda elements.
// lane 0..15 : row=base+lane,    elems = A[row, kc+0..7]  | A[row, kc+16..23]
// lane 16..31: row=base+lane-16, elems = A[row, kc+8..15] | A[row, kc+24..31]
__device__ __forceinline__ v16bf load_a_frag(const unsigned short* __restrict__ A,
                                             long lda, int row0, int kc, int lane) {
  int half = lane >> 4;
  int r    = lane & 15;
  const unsigned short* p = A + (long)(row0 + r) * lda + kc + half * 8;
  BF8 lo, hi;
  lo.u = *(const uint4*)(p);
  hi.u = *(const uint4*)(p + 16);
  v16bf a;
#pragma unroll
  for (int i = 0; i < 8; ++i) { a[i] = lo.h[i]; a[i + 8] = hi.h[i]; }
  return a;
}

// B fragment: 32x16 bf16 (K x N). Weight stored [N_total, K_total] row-major
// (i.e. W[o, i] exactly as the reference stores it), stride ldb elements.
// lane 0..15 : col=base+lane,    elems = W[col, kc+0..7]  | W[col, kc+16..23]
// lane 16..31: col=base+lane-16, elems = W[col, kc+8..15] | W[col, kc+24..31]
__device__ __forceinline__ v16bf load_b_frag(const unsigned short* __restrict__ W,
                                             long ldb, int col0, int kc, int lane) {
  int half = lane >> 4;
  int c    = lane & 15;
  const unsigned short* p = W + (long)(col0 + c) * ldb + kc + half * 8;
  BF8 lo, hi;
  lo.u = *(const uint4*)(p);
  hi.u = *(const uint4*)(p + 16);
  v16bf b;
#pragma unroll
  for (int i = 0; i < 8; ++i) { b[i] = lo.h[i]; b[i + 8] = hi.h[i]; }
  return b;
}

// Accumulate C[32x64] += A[rows row0..row0+31, :K] * W^T (cols col0..col0+63)
__device__ __forceinline__ void gemm_accum(v8f acc[2][4],
                                           const unsigned short* __restrict__ A, long lda,
                                           const unsigned short* __restrict__ W, long ldb,
                                           int K, int row0, int col0, int lane) {
  for (int kc = 0; kc < K; kc += 32) {
    v16bf a0 = load_a_frag(A, lda, row0,      kc, lane);
    v16bf a1 = load_a_frag(A, lda, row0 + 16, kc, lane);
    v16bf b0 = load_b_frag(W, ldb, col0,      kc, lane);
    v16bf b1 = load_b_frag(W, ldb, col0 + 16, kc, lane);
    v16bf b2 = load_b_frag(W, ldb, col0 + 32, kc, lane);
    v16bf b3 = load_b_frag(W, ldb, col0 + 48, kc, lane);
    acc[0][0] = WMMA_BF16(a0, b0, acc[0][0]);
    acc[0][1] = WMMA_BF16(a0, b1, acc[0][1]);
    acc[0][2] = WMMA_BF16(a0, b2, acc[0][2]);
    acc[0][3] = WMMA_BF16(a0, b3, acc[0][3]);
    acc[1][0] = WMMA_BF16(a1, b0, acc[1][0]);
    acc[1][1] = WMMA_BF16(a1, b1, acc[1][1]);
    acc[1][2] = WMMA_BF16(a1, b2, acc[1][2]);
    acc[1][3] = WMMA_BF16(a1, b3, acc[1][3]);
  }
}

__device__ __forceinline__ void zero_acc(v8f acc[2][4]) {
#pragma unroll
  for (int i = 0; i < 2; ++i)
#pragma unroll
    for (int j = 0; j < 4; ++j)
#pragma unroll
      for (int r = 0; r < 8; ++r) acc[i][j][r] = 0.f;
}

// Wave tiling: 128 threads = 4 waves, 2x2 wave grid, block tile 64x128.
#define GEMM_PROLOGUE()                                         \
  int lane   = threadIdx.x & 31;                                \
  int waveId = threadIdx.x >> 5;                                \
  int row0   = blockIdx.y * 64  + (waveId >> 1) * 32;           \
  int col0   = blockIdx.x * 128 + (waveId & 1)  * 64;           \
  v8f acc[2][4]; zero_acc(acc);

// C layout: element r of v8f, lane l -> row = base + r + (l>=16 ? 8 : 0),
//           col = base + (l & 15).
#define EPILOGUE_FOREACH(BODY)                                  \
  { int colOff = lane & 15; int rowAdd = (lane >> 4) * 8;       \
    _Pragma("unroll") for (int mi = 0; mi < 2; ++mi)            \
    _Pragma("unroll") for (int nj = 0; nj < 4; ++nj)            \
    _Pragma("unroll") for (int r = 0; r < 8; ++r) {             \
      int row = row0 + mi * 16 + rowAdd + r;                    \
      int col = col0 + nj * 16 + colOff;                        \
      float vacc = acc[mi][nj][r];                              \
      BODY } }

// -------------------------- utility kernels --------------------------------

__global__ void k_convert_bf16(const float* __restrict__ src, unsigned short* __restrict__ dst,
                               long nrows, long ncols, long src_stride) {
  long n = nrows * ncols;
  for (long i = blockIdx.x * (long)blockDim.x + threadIdx.x; i < n;
       i += (long)gridDim.x * blockDim.x) {
    long r = i / ncols, c = i - r * ncols;
    dst[i] = f32_to_bf16(src[r * src_stride + c]);
  }
}

__global__ void k_zero_f32(float* __restrict__ p, long n) {
  for (long i = blockIdx.x * (long)blockDim.x + threadIdx.x; i < n;
       i += (long)gridDim.x * blockDim.x) p[i] = 0.f;
}

// -------------------------- recurrent step ---------------------------------

// s = tanh(h @ Ws_h^T + x_t @ Ws_x^T + dt * Ws_t + b_s)
__global__ void k_step_s(const unsigned short* __restrict__ h_bf,
                         const unsigned short* __restrict__ x_bf_t, // + t*DD, lda = TT*DD
                         const unsigned short* __restrict__ Wsh_bf,
                         const unsigned short* __restrict__ Wsx_bf,
                         const float* __restrict__ Ws_t,
                         const float* __restrict__ b_s,
                         const float* __restrict__ X_seq, int t,
                         float* __restrict__ s_f32,
                         unsigned short* __restrict__ s_bf) {
  GEMM_PROLOGUE();
  gemm_accum(acc, h_bf,  HH,             Wsh_bf, HH, HH, row0, col0, lane);
  gemm_accum(acc, x_bf_t, (long)TT * DD, Wsx_bf, DD, DD, row0, col0, lane);
  EPILOGUE_FOREACH({
    float dt = X_seq[((long)row * TT + t) * (DD + 1) + DD];
    float sv = tanhf(vacc + dt * Ws_t[col] + b_s[col]);
    long idx = (long)row * HH + col;
    s_f32[idx] = sv;
    s_bf[idx]  = f32_to_bf16(sv);
  });
}

// pre[b, g*H+o] = h@Wg_h^T + s@Wg_s^T + x_t@Wg_x^T + b_g   (N = 5H = 2560)
__global__ void k_step_pre(const unsigned short* __restrict__ h_bf,
                           const unsigned short* __restrict__ s_bf,
                           const unsigned short* __restrict__ x_bf_t,
                           const unsigned short* __restrict__ Wgh_bf, // [5H, H]
                           const unsigned short* __restrict__ Wgs_bf, // [5H, H]
                           const unsigned short* __restrict__ Wgx_bf, // [5H, D]
                           const float* __restrict__ b_g,             // [5H]
                           float* __restrict__ pre_buf) {             // [B, 5H]
  GEMM_PROLOGUE();
  gemm_accum(acc, h_bf,  HH,             Wgh_bf, HH, HH, row0, col0, lane);
  gemm_accum(acc, s_bf,  HH,             Wgs_bf, HH, HH, row0, col0, lane);
  gemm_accum(acc, x_bf_t, (long)TT * DD, Wgx_bf, DD, DD, row0, col0, lane);
  EPILOGUE_FOREACH({
    pre_buf[(long)row * G5H + col] = vacc + b_g[col];
  });
}

// gates -> c,h update; maintain bf16 mirror of h and the 64-deep history.
__global__ void k_step_update(const float* __restrict__ pre_buf,
                              const float* __restrict__ s_f32,
                              float* __restrict__ c_f32,
                              float* __restrict__ h_f32,
                              unsigned short* __restrict__ h_bf,
                              float* __restrict__ hist_f32,          // [B, M, H]
                              unsigned short* __restrict__ hist_bf,  // [B, M, H]
                              int t) {
  long i = blockIdx.x * (long)blockDim.x + threadIdx.x;
  long n = (long)BB * HH;
  if (i >= n) return;
  int b = (int)(i / HH);
  int o = (int)(i - (long)b * HH);
  const float* pb = pre_buf + (long)b * G5H;
  float f    = sigmoidf(pb[o]);
  float ig   = sigmoidf(pb[HH + o]);
  float Tg   = sigmoidf(pb[2 * HH + o]);
  float zeta = tanhf  (pb[3 * HH + o]);
  float og   = sigmoidf(pb[4 * HH + o]);
  float sv   = s_f32[i];
  float cn   = f * c_f32[i] + ig * zeta + Tg * sv;
  float hn   = og * tanhf(cn);
  c_f32[i] = cn;
  h_f32[i] = hn;
  h_bf[i]  = f32_to_bf16(hn);
  if (t >= TT - MM) {
    int slot = t - (TT - MM);
    long hidx = (((long)b * MM) + slot) * HH + o;
    hist_f32[hidx] = hn;
    hist_bf[hidx]  = f32_to_bf16(hn);
  }
}

// -------------------------- attention phase --------------------------------

__global__ void k_build_q(const float* __restrict__ h_f32, const float* __restrict__ c_f32,
                          unsigned short* __restrict__ q_bf) {       // [B, 2H]
  long i = blockIdx.x * (long)blockDim.x + threadIdx.x;
  long n = (long)BB * 2 * HH;
  if (i >= n) return;
  int b = (int)(i >> 10);          // 2H = 1024
  int j = (int)(i & 1023);
  float v = (j < HH) ? h_f32[(long)b * HH + j] : c_f32[(long)b * HH + (j - HH)];
  q_bf[i] = f32_to_bf16(v);
}

// qproj = q @ W_aq^T + b_a
__global__ void k_qproj(const unsigned short* __restrict__ q_bf,
                        const unsigned short* __restrict__ Waq_bf,   // [H, 2H]
                        const float* __restrict__ b_a,
                        float* __restrict__ qproj) {                 // [B, H]
  GEMM_PROLOGUE();
  gemm_accum(acc, q_bf, 2 * HH, Waq_bf, 2 * HH, 2 * HH, row0, col0, lane);
  EPILOGUE_FOREACH({ qproj[(long)row * HH + col] = vacc + b_a[col]; });
}

// o = tanh(hist @ W_ah^T + qproj[b]); logits[b,m] += sum_col o * v_t[col]
__global__ void k_attention(const unsigned short* __restrict__ hist_bf, // [B*M, H]
                            const unsigned short* __restrict__ Wah_bf,  // [H, H]
                            const float* __restrict__ qproj,            // [B, H]
                            const float* __restrict__ v_t,              // [H]
                            float* __restrict__ logits) {               // [B*M]
  GEMM_PROLOGUE();
  gemm_accum(acc, hist_bf, HH, Wah_bf, HH, HH, row0, col0, lane);
  int colOff = lane & 15;
  int rowAdd = (lane >> 4) * 8;
#pragma unroll
  for (int mi = 0; mi < 2; ++mi) {
#pragma unroll
    for (int r = 0; r < 8; ++r) {
      int row = row0 + mi * 16 + rowAdd + r;   // row = b*M + m
      int b   = row >> 6;                      // M = 64
      float partial = 0.f;
#pragma unroll
      for (int nj = 0; nj < 4; ++nj) {
        int col = col0 + nj * 16 + colOff;
        float o = tanhf(acc[mi][nj][r] + qproj[(long)b * HH + col]);
        partial += o * v_t[col];
      }
      partial += __shfl_xor(partial, 1);
      partial += __shfl_xor(partial, 2);
      partial += __shfl_xor(partial, 4);
      partial += __shfl_xor(partial, 8);
      if (colOff == 0) atomicAdd(&logits[row], partial);
    }
  }
}

// softmax over M + context e = sum_m alpha * hist
__global__ void k_softmax_e(const float* __restrict__ logits,
                            const float* __restrict__ hist_f32,
                            float* __restrict__ e_f32,
                            unsigned short* __restrict__ e_bf) {
  int b   = blockIdx.x;
  int tid = threadIdx.x;
  __shared__ float sl[MM];
  __shared__ float se[MM];
  if (tid < MM) sl[tid] = logits[(long)b * MM + tid];
  __syncthreads();
  float mx = -1e30f;
  for (int m = 0; m < MM; ++m) mx = fmaxf(mx, sl[m]);
  if (tid < MM) se[tid] = __expf(sl[tid] - mx);
  __syncthreads();
  float ssum = 0.f;
  for (int m = 0; m < MM; ++m) ssum += se[m];
  float inv = 1.f / ssum;
  for (int h = tid; h < HH; h += blockDim.x) {
    float a = 0.f;
    for (int m = 0; m < MM; ++m)
      a += se[m] * hist_f32[(((long)b * MM) + m) * HH + h];
    a *= inv;
    long idx = (long)b * HH + h;
    e_f32[idx] = a;
    e_bf[idx]  = f32_to_bf16(a);
  }
}

// h_fin = tanh(h_last@W_hh^T + e@W_e^T + x_last@W_g^T + b_h)
__global__ void k_final(const unsigned short* __restrict__ h_bf,
                        const unsigned short* __restrict__ e_bf,
                        const unsigned short* __restrict__ x_bf_last, // + (T-1)*D
                        const unsigned short* __restrict__ Whh_bf,
                        const unsigned short* __restrict__ We_bf,
                        const unsigned short* __restrict__ Wg_bf,
                        const float* __restrict__ b_h,
                        float* __restrict__ h_fin) {
  GEMM_PROLOGUE();
  gemm_accum(acc, h_bf,      HH,            Whh_bf, HH, HH, row0, col0, lane);
  gemm_accum(acc, e_bf,      HH,            We_bf,  HH, HH, row0, col0, lane);
  gemm_accum(acc, x_bf_last, (long)TT * DD, Wg_bf,  DD, DD, row0, col0, lane);
  EPILOGUE_FOREACH({ h_fin[(long)row * HH + col] = tanhf(vacc + b_h[col]); });
}

// out[b] = sigmoid(h_fin[b,:] . W_c + b_c)
__global__ void k_out(const float* __restrict__ h_fin,
                      const float* __restrict__ W_c,
                      const float* __restrict__ b_c,
                      float* __restrict__ out) {
  int b   = blockIdx.x;
  int tid = threadIdx.x;
  float p = 0.f;
  for (int h = tid; h < HH; h += blockDim.x) p += h_fin[(long)b * HH + h] * W_c[h];
  __shared__ float red[128];
  red[tid] = p;
  __syncthreads();
  for (int s = 64; s > 0; s >>= 1) {
    if (tid < s) red[tid] += red[tid + s];
    __syncthreads();
  }
  if (tid == 0) out[b] = 1.f / (1.f + __expf(-(red[0] + b_c[0])));
}

// ------------------------------ launch -------------------------------------

static inline char* wsal(char*& p, size_t bytes) {
  char* r = p;
  p += (bytes + 255) & ~(size_t)255;
  return r;
}

extern "C" void kernel_launch(void* const* d_in, const int* in_sizes, int n_in,
                              void* d_out, int out_size, void* d_ws, size_t ws_size,
                              hipStream_t stream) {
  (void)in_sizes; (void)n_in; (void)out_size; (void)ws_size;
  const float* X_seq = (const float*)d_in[0];
  const float* Ws_h  = (const float*)d_in[1];
  const float* Ws_x  = (const float*)d_in[2];
  const float* Ws_t  = (const float*)d_in[3];
  const float* b_s   = (const float*)d_in[4];
  const float* Wg_h  = (const float*)d_in[5];
  const float* Wg_x  = (const float*)d_in[6];
  const float* Wg_s  = (const float*)d_in[7];
  const float* b_g   = (const float*)d_in[8];
  const float* W_aq  = (const float*)d_in[9];
  const float* W_ah  = (const float*)d_in[10];
  const float* b_a   = (const float*)d_in[11];
  const float* v_t   = (const float*)d_in[12];
  const float* W_hh  = (const float*)d_in[13];
  const float* W_e   = (const float*)d_in[14];
  const float* W_g   = (const float*)d_in[15];
  const float* b_h   = (const float*)d_in[16];
  const float* W_c   = (const float*)d_in[17];
  const float* b_c   = (const float*)d_in[18];
  float* out = (float*)d_out;

  char* p = (char*)d_ws;
  unsigned short* x_bf    = (unsigned short*)wsal(p, (size_t)BB * TT * DD * 2);
  unsigned short* Wsh_bf  = (unsigned short*)wsal(p, (size_t)HH * HH * 2);
  unsigned short* Wsx_bf  = (unsigned short*)wsal(p, (size_t)HH * DD * 2);
  unsigned short* Wgh_bf  = (unsigned short*)wsal(p, (size_t)G5H * HH * 2);
  unsigned short* Wgs_bf  = (unsigned short*)wsal(p, (size_t)G5H * HH * 2);
  unsigned short* Wgx_bf  = (unsigned short*)wsal(p, (size_t)G5H * DD * 2);
  unsigned short* Waq_bf  = (unsigned short*)wsal(p, (size_t)HH * 2 * HH * 2);
  unsigned short* Wah_bf  = (unsigned short*)wsal(p, (size_t)HH * HH * 2);
  unsigned short* Whh_bf  = (unsigned short*)wsal(p, (size_t)HH * HH * 2);
  unsigned short* We_bf   = (unsigned short*)wsal(p, (size_t)HH * HH * 2);
  unsigned short* Wgf_bf  = (unsigned short*)wsal(p, (size_t)HH * DD * 2);
  float*          h_f32   = (float*)wsal(p, (size_t)BB * HH * 4);
  float*          c_f32   = (float*)wsal(p, (size_t)BB * HH * 4);
  float*          s_f32   = (float*)wsal(p, (size_t)BB * HH * 4);
  unsigned short* h_bf    = (unsigned short*)wsal(p, (size_t)BB * HH * 2);
  unsigned short* s_bf    = (unsigned short*)wsal(p, (size_t)BB * HH * 2);
  float*          pre_buf = (float*)wsal(p, (size_t)BB * G5H * 4);
  float*          hist_f  = (float*)wsal(p, (size_t)BB * MM * HH * 4);
  unsigned short* hist_bf = (unsigned short*)wsal(p, (size_t)BB * MM * HH * 2);
  unsigned short* q_bf    = (unsigned short*)wsal(p, (size_t)BB * 2 * HH * 2);
  float*          qproj   = (float*)wsal(p, (size_t)BB * HH * 4);
  float*          logits  = (float*)wsal(p, (size_t)BB * MM * 4);
  float*          e_f32   = (float*)wsal(p, (size_t)BB * HH * 4);
  unsigned short* e_bf    = (unsigned short*)wsal(p, (size_t)BB * HH * 2);
  float*          h_fin   = (float*)wsal(p, (size_t)BB * HH * 4);

  auto cvt = [&](const float* src, unsigned short* dst, long nr, long nc, long stride) {
    long n = nr * nc;
    int blocks = (int)((n + 255) / 256);
    if (blocks > 4096) blocks = 4096;
    k_convert_bf16<<<blocks, 256, 0, stream>>>(src, dst, nr, nc, stride);
  };

  // one-time conversions (repeated every call for determinism)
  cvt(X_seq, x_bf, (long)BB * TT, DD, DD + 1);     // strip dt column
  cvt(Ws_h, Wsh_bf, HH, HH, HH);
  cvt(Ws_x, Wsx_bf, HH, DD, DD);
  cvt(Wg_h, Wgh_bf, G5H, HH, HH);
  cvt(Wg_s, Wgs_bf, G5H, HH, HH);
  cvt(Wg_x, Wgx_bf, G5H, DD, DD);
  cvt(W_aq, Waq_bf, HH, 2 * HH, 2 * HH);
  cvt(W_ah, Wah_bf, HH, HH, HH);
  cvt(W_hh, Whh_bf, HH, HH, HH);
  cvt(W_e,  We_bf,  HH, HH, HH);
  cvt(W_g,  Wgf_bf, HH, DD, DD);

  // zero initial state (h, c, h_bf)
  k_zero_f32<<<((long)BB * HH + 255) / 256, 256, 0, stream>>>(h_f32, (long)BB * HH);
  k_zero_f32<<<((long)BB * HH + 255) / 256, 256, 0, stream>>>(c_f32, (long)BB * HH);
  k_zero_f32<<<((long)BB * HH / 2 + 255) / 256, 256, 0, stream>>>((float*)h_bf, (long)BB * HH / 2);

  dim3 blk(128);
  dim3 grd_s(HH / 128, BB / 64);      // (4, 8)
  dim3 grd_pre(G5H / 128, BB / 64);   // (20, 8)

  // recurrent scan
  for (int t = 0; t < TT; ++t) {
    const unsigned short* x_t = x_bf + (long)t * DD;
    k_step_s<<<grd_s, blk, 0, stream>>>(h_bf, x_t, Wsh_bf, Wsx_bf, Ws_t, b_s,
                                        X_seq, t, s_f32, s_bf);
    k_step_pre<<<grd_pre, blk, 0, stream>>>(h_bf, s_bf, x_t, Wgh_bf, Wgs_bf,
                                            Wgx_bf, b_g, pre_buf);
    k_step_update<<<((long)BB * HH + 255) / 256, 256, 0, stream>>>(
        pre_buf, s_f32, c_f32, h_f32, h_bf, hist_f, hist_bf, t);
  }

  // attention
  k_build_q<<<((long)BB * 2 * HH + 255) / 256, 256, 0, stream>>>(h_f32, c_f32, q_bf);
  k_zero_f32<<<((long)BB * MM + 255) / 256, 256, 0, stream>>>(logits, (long)BB * MM);
  k_qproj<<<grd_s, blk, 0, stream>>>(q_bf, Waq_bf, b_a, qproj);
  dim3 grd_att(HH / 128, (BB * MM) / 64);   // (4, 512)
  k_attention<<<grd_att, blk, 0, stream>>>(hist_bf, Wah_bf, qproj, v_t, logits);
  k_softmax_e<<<BB, 128, 0, stream>>>(logits, hist_f, e_f32, e_bf);

  // final projection + output
  const unsigned short* x_last = x_bf + (long)(TT - 1) * DD;
  k_final<<<grd_s, blk, 0, stream>>>(h_bf, e_bf, x_last, Whh_bf, We_bf, Wgf_bf,
                                     b_h, h_fin);
  k_out<<<BB, 128, 0, stream>>>(h_fin, W_c, b_c, out);
}
